// Attention_26371099197862
// MI455X (gfx1250) — matrix operand order repaired and checked by reading, not compiled, not run on hardware
//
#include <hip/hip_runtime.h>

// ---------------------------------------------------------------------------
// CDNA5 / gfx1250 bf16 WMMA attention pipeline (wave32).
// Flash attention computed transposed (S^T = K Q^T, O^T = V^T P^T) so softmax
// stats are in-lane reductions; the only cross-lane traffic is the fixed
// lane^16 pattern, done with v_permlanex16_b32 (pure VALU, no LDS waits).
// ---------------------------------------------------------------------------

typedef __attribute__((ext_vector_type(16))) __bf16 v16bf;
typedef __attribute__((ext_vector_type(8)))  __bf16 v8bf;
typedef __attribute__((ext_vector_type(8)))  float  v8f;
typedef __attribute__((ext_vector_type(4)))  int    v4i;

#define BB 4
#define N1C 2048
#define N2C 2048
#define DXC 128
#define DVC 512
#define DKC 256
#define HC  8
#define HSC 64
#define ROWS 8192          // B*N1 == B*N2

// ---------------- cross-half (lane ^ 16) exchange, VALU path ----------------

__device__ inline unsigned xswap_u32(unsigned u) {
#if __has_builtin(__builtin_amdgcn_permlanex16)
  // identity selects: lane i <-> lane i+16
  return __builtin_amdgcn_permlanex16(u, u, 0x76543210u, 0xfedcba98u,
                                      false, false);
#else
  return (unsigned)__shfl_xor((int)u, 16);
#endif
}
__device__ inline float xswap_f32(float v) {
  return __builtin_bit_cast(float, xswap_u32(__builtin_bit_cast(unsigned, v)));
}
__device__ inline int xswap_i32(int v) {
  return (int)xswap_u32((unsigned)v);
}

// ---------------- WMMA helpers (contiguous k order per lane half) ----------------

__device__ inline v8f wmma_bf16(v16bf a, v16bf b, v8f c) {
  return __builtin_amdgcn_wmma_f32_16x16x32_bf16(false, a, false, b,
                                                 (short)0, c, false, false);
}

// A fragment 16x32 (M x K), A row-major [*, lda]:
//  lane<16 : elems 0..15 = K k0..k0+15
//  lane>=16: elems 0..15 = K k0+16..k0+31
__device__ inline v16bf load_a_frag(const __bf16* base, int ld, int m0, int k0,
                                    int lane) {
  int row  = m0 + (lane & 15);
  int half = lane >> 4;
  const __bf16* p = base + (long)row * ld + k0 + half * 16;
  v8bf lo = *(const v8bf*)p;
  v8bf hi = *(const v8bf*)(p + 8);
  return __builtin_shufflevector(lo, hi, 0, 1, 2, 3, 4, 5, 6, 7,
                                 8, 9, 10, 11, 12, 13, 14, 15);
}

// B fragment 32x16 (K x N) from BT (i.e. B transposed, [N][K] row-major):
//  lane<16 : col = n0+lane, elems = K k0..k0+15
//  lane>=16: col = n0+lane-16, elems = K k0+16..k0+31
__device__ inline v16bf load_b_frag(const __bf16* baseT, int ld, int n0, int k0,
                                    int lane) {
  int col  = n0 + (lane & 15);
  int half = lane >> 4;
  return *(const v16bf*)(baseT + (long)col * ld + k0 + half * 16);
}

// ---------------- converters ------------------------------------------------

__global__ void f2bf_kernel(const float* __restrict__ in, __bf16* __restrict__ out,
                            long n) {
  long i = (long)blockIdx.x * blockDim.x + threadIdx.x;
  if (i < n) out[i] = (__bf16)in[i];
}

// out[z][c][r] = in[z][r][c]  (weight -> BT layout, fp32 -> bf16)
__global__ void transpose_f2bf_kernel(const float* __restrict__ in,
                                      __bf16* __restrict__ out,
                                      int rows, int cols) {
  long base = (long)blockIdx.z * rows * cols;
  int i = blockIdx.x * blockDim.x + threadIdx.x;
  if (i < rows * cols) {
    int r = i / cols, c = i % cols;
    out[base + (long)c * rows + r] = (__bf16)in[base + i];
  }
}

// WoRepT[f][k] = Wo[k % 64][f]  -> final GEMM's BT, K = H*HS = 512
__global__ void build_worep_kernel(const float* __restrict__ Wo,
                                   __bf16* __restrict__ out) {
  int i = blockIdx.x * blockDim.x + threadIdx.x;
  if (i < 512 * 512) {
    int f = i / 512, k = i % 512;
    out[i] = (__bf16)Wo[(long)(k & 63) * 512 + f];
  }
}

// ---------------- generic batched bf16 WMMA GEMM ---------------------------
// C[m][n] = sum_k A[m][k] * BT[n][k] + biasScale*bias[n], optional ReLU,
// optional transposed store (C[n][m]), optional f32 output.
// Block = 4 waves; wave tile = 16x64; block tile = 64x64.

template <bool RELU, bool TRANSOUT, bool F32OUT>
__global__ void gemm_bf16_wmma(const __bf16* __restrict__ A, int lda, long aBs,
                               const __bf16* __restrict__ BT, int ldbt, long bBs,
                               const float* __restrict__ bias, long biasBs,
                               float biasScale,
                               void* __restrict__ Cout, int ldc, long cBs,
                               int M, int N, int K) {
  int lane = threadIdx.x & 31;
  int w    = threadIdx.x >> 5;
  int z    = blockIdx.z;
  A    += (long)z * aBs;
  BT   += (long)z * bBs;
  bias += (long)z * biasBs;
  __bf16* Cb = (__bf16*)Cout + (long)z * cBs;
  float*  Cf = (float*)Cout + (long)z * cBs;

  int m0 = blockIdx.y * 64 + w * 16;   // wave-uniform
  int n0 = blockIdx.x * 64;
  if (m0 >= M || n0 >= N) return;

  v8f acc[4] = {};
  for (int k0 = 0; k0 < K; k0 += 32) {
    v16bf a = load_a_frag(A, lda, m0, k0, lane);
    if (k0 + 32 < K)  // prefetch next A K-slice (global_prefetch_b8)
      __builtin_prefetch(A + (long)(m0 + (lane & 15)) * lda + k0 + 32, 0, 1);
#pragma unroll
    for (int t = 0; t < 4; t++) {
      v16bf b = load_b_frag(BT, ldbt, n0 + t * 16, k0, lane);
      acc[t] = wmma_bf16(a, b, acc[t]);
    }
  }

  int nl = lane & 15, hi = lane >> 4;
#pragma unroll
  for (int t = 0; t < 4; t++) {
    int col = n0 + t * 16 + nl;
    float bv = biasScale * bias[col];
#pragma unroll
    for (int i = 0; i < 8; i++) {
      int row = m0 + i + 8 * hi;
      float v = acc[t][i] + bv;
      if (RELU) v = fmaxf(v, 0.0f);
      long idx = TRANSOUT ? ((long)col * ldc + row) : ((long)row * ldc + col);
      if (F32OUT) Cf[idx] = v; else Cb[idx] = (__bf16)v;
    }
  }
}

// ---------------- flash attention (transposed formulation) ------------------
// qh : [H][ROWS][64]   (row = b*N2 + m)  -> B-fragment of S^T (loop invariant)
// kh : [H][ROWS][64]   (row = b*N1 + n)  -> A-fragment of S^T
// vhT: [H][64][ROWS]   (col = b*N1 + n)  -> A-fragment of O^T
// ocat: [ROWS][512]    (col = h*64 + e)   head-concatenated output
// One wave = one 16-col m tile of S^T; block = 4 waves; online softmax over N1.
// C-fragment of S^T holds one m-column per lane => per-lane scalar stats.

__global__ void flash_attn_kernel(const __bf16* __restrict__ qh,
                                  const __bf16* __restrict__ kh,
                                  const __bf16* __restrict__ vhT,
                                  __bf16* __restrict__ ocat) {
  int lane = threadIdx.x & 31;
  int w    = threadIdx.x >> 5;
  int nl   = lane & 15, hi = lane >> 4;
  int b = blockIdx.y / HC;
  int h = blockIdx.y % HC;
  int m0 = blockIdx.x * 64 + w * 16;

  const __bf16* qbase = qh + (long)h * ROWS * HSC;
  const __bf16* kbase = kh + (long)h * ROWS * HSC + (long)b * N1C * HSC;
  const __bf16* vbase = vhT + (long)h * HSC * ROWS;  // [64][ROWS]
  int qrow0 = b * N2C + m0;
  int vcol0 = b * N1C;

  // Q as B-fragments for the full HS=64 K-extent (loop invariant)
  v16bf bq0 = load_b_frag(qbase, HSC, qrow0, 0, lane);
  v16bf bq1 = load_b_frag(qbase, HSC, qrow0, 32, lane);

  float rmax = -3.0e38f, rsum = 0.0f;   // per-lane: column m = m0 + nl
  v8f o[4] = {};                        // O^T tiles: rows e, col m per lane
  const float SCALE = 0.125f;           // 1/sqrt(HS)

  for (int n0 = 0; n0 < N1C; n0 += 32) {
    // S^T[32n x 16m] = kh(32x64) @ qh^T(64x16), two 16x16 C tiles over n
    v16bf ak00 = load_a_frag(kbase, HSC, n0,      0,  lane);
    v16bf ak01 = load_a_frag(kbase, HSC, n0,      32, lane);
    v16bf ak10 = load_a_frag(kbase, HSC, n0 + 16, 0,  lane);
    v16bf ak11 = load_a_frag(kbase, HSC, n0 + 16, 32, lane);
    if (n0 + 32 < N1C)
      __builtin_prefetch(kbase + (long)(n0 + 32 + (lane & 15)) * HSC, 0, 1);
    v8f st0 = {}, st1 = {};
    st0 = wmma_bf16(ak00, bq0, st0);
    st0 = wmma_bf16(ak01, bq1, st0);
    st1 = wmma_bf16(ak10, bq0, st1);
    st1 = wmma_bf16(ak11, bq1, st1);

    // online softmax: reduce over n = 16 regs in-lane + one lane^16 combine
    float a0[8], a1[8];
    float pmax = -3.0e38f;
#pragma unroll
    for (int i = 0; i < 8; i++) {
      a0[i] = st0[i] * SCALE;
      a1[i] = st1[i] * SCALE;
      pmax = fmaxf(pmax, fmaxf(a0[i], a1[i]));
    }
    pmax = fmaxf(pmax, xswap_f32(pmax));
    float nm    = fmaxf(rmax, pmax);
    float alpha = __expf(rmax - nm);
    rmax = nm;

    v8bf own0, own1;           // bf16 P^T halves held by this lane
    float psum = 0.0f;
#pragma unroll
    for (int i = 0; i < 8; i++) {
      float p0 = __expf(a0[i] - nm);
      float p1 = __expf(a1[i] - nm);
      psum += p0 + p1;
      own0[i] = (__bf16)p0;
      own1[i] = (__bf16)p1;
    }
    rsum = rsum * alpha + psum + xswap_f32(psum);
#pragma unroll
    for (int t = 0; t < 4; t++)
#pragma unroll
      for (int i = 0; i < 8; i++) o[t][i] *= alpha;

    // assemble P^T B-fragment: lane^16 exchange of the half the partner needs.
    // B-frag (K=32): lane hi=0 wants n 0..15 = [own0 | partner own0],
    //                lane hi=1 wants n 16..31 = [partner own1 | own1].
    v8bf sendv = hi ? own0 : own1;   // send what the partner needs
    v4i  si = __builtin_bit_cast(v4i, sendv);
#pragma unroll
    for (int j = 0; j < 4; j++) si[j] = xswap_i32(si[j]);
    v8bf recv = __builtin_bit_cast(v8bf, si);
    v16bf pB = __builtin_shufflevector(hi ? recv : own0, hi ? own1 : recv,
                                       0, 1, 2, 3, 4, 5, 6, 7,
                                       8, 9, 10, 11, 12, 13, 14, 15);

    // O^T[64e x 16m] += V^T(16x32 per e-tile) @ P^T(32x16)
#pragma unroll
    for (int t = 0; t < 4; t++) {
      v16bf aV = load_a_frag(vbase, ROWS, t * 16, vcol0 + n0, lane);
      o[t] = wmma_bf16(aV, pB, o[t]);
    }
  }

  // normalize, store: per lane one output row (m), 4x contiguous 16B stores
  float inv = 1.0f / rsum;
  __bf16* orow = ocat + (long)(qrow0 + nl) * 512 + h * 64;
#pragma unroll
  for (int t = 0; t < 4; t++) {
    v8bf ov;
#pragma unroll
    for (int i = 0; i < 8; i++) ov[i] = (__bf16)(o[t][i] * inv);
    *(v8bf*)(orow + t * 16 + 8 * hi) = ov;   // e = t*16 + 8*hi + i
  }
}

// ---------------------------------------------------------------------------

extern "C" void kernel_launch(void* const* d_in, const int* in_sizes, int n_in,
                              void* d_out, int out_size, void* d_ws, size_t ws_size,
                              hipStream_t stream) {
  (void)in_sizes; (void)n_in; (void)out_size; (void)ws_size;

  const float* ctx  = (const float*)d_in[0];   // [B,N1,DX]
  const float* tgt  = (const float*)d_in[1];   // [B,N2,DX]
  const float* r    = (const float*)d_in[2];   // [B,N1,DV]
  const float* W1   = (const float*)d_in[3];   // [DX,256]
  const float* b1   = (const float*)d_in[4];   // [256]
  const float* W2   = (const float*)d_in[5];   // [256,DK]
  const float* b2   = (const float*)d_in[6];   // [DK]
  const float* Wq   = (const float*)d_in[7];   // [H,DK,HS]
  const float* bq   = (const float*)d_in[8];   // [H,HS]
  const float* Wk   = (const float*)d_in[9];
  const float* bk   = (const float*)d_in[10];
  const float* Wv   = (const float*)d_in[11];  // [H,DV,HS]
  const float* bv   = (const float*)d_in[12];
  const float* Wo   = (const float*)d_in[13];  // [HS,DV]
  const float* bo   = (const float*)d_in[14];  // [DV]
  float* out = (float*)d_out;                  // [B,N2,DV]

  // workspace carve-up (bf16 elements, 256B aligned)
  char* ws = (char*)d_ws;
  size_t off = 0;
  auto take = [&](size_t elems) -> __bf16* {
    __bf16* p = (__bf16*)(ws + off);
    off = (off + elems * sizeof(__bf16) + 255) & ~(size_t)255;
    return p;
  };
  __bf16* cxbf = take((size_t)ROWS * DXC);
  __bf16* txbf = take((size_t)ROWS * DXC);
  __bf16* rbf  = take((size_t)ROWS * DVC);
  __bf16* W1T  = take((size_t)256 * DXC);
  __bf16* W2T  = take((size_t)DKC * 256);
  __bf16* WqT  = take((size_t)HC * HSC * DKC);
  __bf16* WkT  = take((size_t)HC * HSC * DKC);
  __bf16* WvT  = take((size_t)HC * HSC * DVC);
  __bf16* WoRT = take((size_t)512 * 512);
  __bf16* hK   = take((size_t)ROWS * 256);
  __bf16* hQ   = take((size_t)ROWS * 256);
  __bf16* kbf  = take((size_t)ROWS * DKC);
  __bf16* qbf  = take((size_t)ROWS * DKC);
  __bf16* qhp  = take((size_t)HC * ROWS * HSC);
  __bf16* khp  = take((size_t)HC * ROWS * HSC);
  __bf16* vhT  = take((size_t)HC * HSC * ROWS);
  __bf16* ocat = take((size_t)ROWS * 512);

  const int T = 256;
  // activation converts
  f2bf_kernel<<<(ROWS * DXC + T - 1) / T, T, 0, stream>>>(ctx, cxbf, (long)ROWS * DXC);
  f2bf_kernel<<<(ROWS * DXC + T - 1) / T, T, 0, stream>>>(tgt, txbf, (long)ROWS * DXC);
  f2bf_kernel<<<(ROWS * DVC + T - 1) / T, T, 0, stream>>>(r, rbf, (long)ROWS * DVC);
  // weight transpose-converts (BT layouts)
  transpose_f2bf_kernel<<<dim3((DXC * 256 + T - 1) / T, 1, 1), T, 0, stream>>>(W1, W1T, DXC, 256);
  transpose_f2bf_kernel<<<dim3((256 * DKC + T - 1) / T, 1, 1), T, 0, stream>>>(W2, W2T, 256, DKC);
  transpose_f2bf_kernel<<<dim3((DKC * HSC + T - 1) / T, 1, HC), T, 0, stream>>>(Wq, WqT, DKC, HSC);
  transpose_f2bf_kernel<<<dim3((DKC * HSC + T - 1) / T, 1, HC), T, 0, stream>>>(Wk, WkT, DKC, HSC);
  transpose_f2bf_kernel<<<dim3((DVC * HSC + T - 1) / T, 1, HC), T, 0, stream>>>(Wv, WvT, DVC, HSC);
  build_worep_kernel<<<(512 * 512 + T - 1) / T, T, 0, stream>>>(Wo, WoRT);

  dim3 blk(128);
  // shared MLP: h = relu(x@W1+b1); kq = h@W2+b2
  gemm_bf16_wmma<true, false, false><<<dim3(256 / 64, ROWS / 64, 1), blk, 0, stream>>>(
      cxbf, DXC, 0, W1T, DXC, 0, b1, 0, 1.0f, hK, 256, 0, ROWS, 256, DXC);
  gemm_bf16_wmma<true, false, false><<<dim3(256 / 64, ROWS / 64, 1), blk, 0, stream>>>(
      txbf, DXC, 0, W1T, DXC, 0, b1, 0, 1.0f, hQ, 256, 0, ROWS, 256, DXC);
  gemm_bf16_wmma<false, false, false><<<dim3(DKC / 64, ROWS / 64, 1), blk, 0, stream>>>(
      hK, 256, 0, W2T, 256, 0, b2, 0, 1.0f, kbf, DKC, 0, ROWS, DKC, 256);
  gemm_bf16_wmma<false, false, false><<<dim3(DKC / 64, ROWS / 64, 1), blk, 0, stream>>>(
      hQ, 256, 0, W2T, 256, 0, b2, 0, 1.0f, qbf, DKC, 0, ROWS, DKC, 256);
  // per-head projections (batched over heads via blockIdx.z)
  gemm_bf16_wmma<false, false, false><<<dim3(1, ROWS / 64, HC), blk, 0, stream>>>(
      qbf, DKC, 0, WqT, DKC, (long)HSC * DKC, bq, HSC, 1.0f,
      qhp, HSC, (long)ROWS * HSC, ROWS, HSC, DKC);
  gemm_bf16_wmma<false, false, false><<<dim3(1, ROWS / 64, HC), blk, 0, stream>>>(
      kbf, DKC, 0, WkT, DKC, (long)HSC * DKC, bk, HSC, 1.0f,
      khp, HSC, (long)ROWS * HSC, ROWS, HSC, DKC);
  // V projection stored transposed: vhT[h][e][row]
  gemm_bf16_wmma<false, true, false><<<dim3(1, ROWS / 64, HC), blk, 0, stream>>>(
      rbf, DVC, 0, WvT, DVC, (long)HSC * DVC, bv, HSC, 1.0f,
      vhT, ROWS, (long)HSC * ROWS, ROWS, HSC, DVC);
  // flash attention
  flash_attn_kernel<<<dim3(N2C / 64, BB * HC), blk, 0, stream>>>(qhp, khp, vhT, ocat);
  // rep = ocat @ WoRep + H*bo  (f32 out)
  gemm_bf16_wmma<false, false, true><<<dim3(512 / 64, ROWS / 64, 1), blk, 0, stream>>>(
      ocat, 512, 0, WoRT, 512, 0, bo, 0, (float)HC, out, 512, 0, ROWS, 512, 512);
}
